// HyperbolicMultiheadAttention_2216203125252
// MI455X (gfx1250) — matrix operand order repaired and checked by reading, not compiled, not run on hardware
//
#include <hip/hip_runtime.h>
#include <hip/hip_bf16.h>
#include <math.h>

// ---------------------------------------------------------------------------
// Hyperbolic multi-head attention for MI455X (gfx1250, wave32, WMMA).
// B=8, S=1024, E=768, H=12, D=64.  All matmuls via v_wmma_f32_16x16x32_f16.
// Operands pre-converted to f16 so the GEMM inner loops are pure
// load_b128 + wmma (no v_cvt chains).
// ---------------------------------------------------------------------------

typedef __attribute__((ext_vector_type(16))) _Float16 v16h;
typedef __attribute__((ext_vector_type(8)))  _Float16 v8h;
typedef __attribute__((ext_vector_type(8)))  float    v8f;

#define EPSF 1e-15f
static constexpr int BB = 8, SS = 1024, EE = 768, HH = 12, DD = 64;

// ---- fragment builders (CDNA5 16-bit WMMA VGPR layouts) -------------------
// A (16x32 f16): lane m(0-15) row M=m, halves 0..7 -> K=off..off+7,
//                halves 8..15 -> K=off+16..off+23; lanes 16-31: +8 K offset.
// B (32x16 f16): lane n(0-15) col N=n, same K layout as A (K=0..7,16..23; lanes 16-31 +8).
__device__ inline v16h frag_from_h(const _Float16* p0, const _Float16* p1) {
  v8h a = *(const v8h*)p0, b = *(const v8h*)p1;
  v16h r;
#pragma unroll
  for (int i = 0; i < 8; i++) { r[i] = a[i]; r[8 + i] = b[i]; }
  return r;
}

__device__ inline float red_sum16(float v) {
#pragma unroll
  for (int m = 8; m >= 1; m >>= 1) v += __shfl_xor(v, m, 32);
  return v;
}
__device__ inline float red_sum32(float v) {
#pragma unroll
  for (int m = 16; m >= 1; m >>= 1) v += __shfl_xor(v, m, 32);
  return v;
}
__device__ inline float red_max32(float v) {
#pragma unroll
  for (int m = 16; m >= 1; m >>= 1) v = fmaxf(v, __shfl_xor(v, m, 32));
  return v;
}

// ---- fp32 -> f16 bulk conversion (done once per tensor) -------------------
__global__ void f32_to_f16_kernel(const float* __restrict__ in,
                                  _Float16* __restrict__ out, int n) {
  const int i = (blockIdx.x * blockDim.x + threadIdx.x) * 8;
  if (i + 8 <= n) {
    float4 a = *(const float4*)(in + i);
    float4 b = *(const float4*)(in + i + 4);
    v8h r;
    r[0] = (_Float16)a.x; r[1] = (_Float16)a.y; r[2] = (_Float16)a.z; r[3] = (_Float16)a.w;
    r[4] = (_Float16)b.x; r[5] = (_Float16)b.y; r[6] = (_Float16)b.z; r[7] = (_Float16)b.w;
    *(v8h*)(out + i) = r;
  }
}

// ---- row L2 norms (clamped at EPS), one wave per row ----------------------
__global__ void rownorm_kernel(const float* __restrict__ X, float* __restrict__ out) {
  const int row = blockIdx.x, lane = threadIdx.x;
  const float* p = X + (size_t)row * EE;
  float s = 0.f;
  for (int i = lane; i < EE; i += 32) { float v = p[i]; s += v * v; }
  s = red_sum32(s);
  if (lane == 0) out[row] = fmaxf(sqrtf(s), EPSF);
}

// ---- sum of squares of a length-E vector (bias^2 scalar) ------------------
__global__ void sqsum_kernel(const float* __restrict__ v, float* __restrict__ out) {
  const int lane = threadIdx.x;
  float s = 0.f;
  for (int i = lane; i < EE; i += 32) { float t = v[i]; s += t * t; }
  s = red_sum32(s);
  if (lane == 0) out[0] = s;
}

// ---------------------------------------------------------------------------
// hyp_proj: out_row = mobius_add(mobius_matvec(W, x_row), bias)
// Block: 16 rows x 768 cols.  8 waves x 6 N-tiles x 24 K-steps of WMMA.
// X, W pre-converted to f16 (row-major).
// mode 0: Q/K -> clamp min(o, 1-1e-5), store f16 [B,H,S,D], write per-head
//         sum-of-squares to sqout [B,H,S].
// mode 1: V  -> store f16 transposed [B,H,D,S].
// mode 2: O  -> store f32 [B,S,E] (final output).
// ---------------------------------------------------------------------------
__global__ __launch_bounds__(256) void hyp_proj_kernel(
    const _Float16* __restrict__ Xh, const _Float16* __restrict__ Wh,
    const float* __restrict__ bias, const float* __restrict__ xnorm,
    const float* __restrict__ b2p, int mode,
    _Float16* __restrict__ outh, float* __restrict__ sqout,
    float* __restrict__ outf) {
  __shared__ float s_rowsum[16];
  __shared__ float s_rowdot[16];
  __shared__ float s_sq[16 * 12];

  const int tid = threadIdx.x, wid = tid >> 5, lane = tid & 31;
  const int nl = lane & 15, hi = (lane >> 4) & 1;
  const int row0 = blockIdx.x * 16;

  if (tid < 16) { s_rowsum[tid] = 0.f; s_rowdot[tid] = 0.f; }
  if (tid < 192) s_sq[tid] = 0.f;
  __syncthreads();

  const _Float16* xrow = Xh + (size_t)(row0 + nl) * EE + (hi ? 8 : 0);
  const _Float16* wr[6];
#pragma unroll
  for (int t = 0; t < 6; t++)
    wr[t] = Wh + (size_t)((wid * 6 + t) * 16 + nl) * EE + (hi ? 8 : 0);

  v8f acc[6];
#pragma unroll
  for (int t = 0; t < 6; t++) acc[t] = {};

#pragma unroll 2
  for (int kb = 0; kb < EE; kb += 32) {
    __builtin_prefetch(xrow + kb + 128, 0, 0);  // global_prefetch hint
    v16h A = frag_from_h(xrow + kb, xrow + kb + 16);
    v16h Bf[6];
#pragma unroll
    for (int t = 0; t < 6; t++) Bf[t] = frag_from_h(wr[t] + kb, wr[t] + kb + 16);
#pragma unroll
    for (int t = 0; t < 6; t++)
      acc[t] = __builtin_amdgcn_wmma_f32_16x16x32_f16(
          false, A, false, Bf[t], (short)0, acc[t], false, false);
  }

  // bias value for this lane's column in each owned tile
  float bcol[6];
#pragma unroll
  for (int t = 0; t < 6; t++) bcol[t] = bias[(wid * 6 + t) * 16 + nl];

  // row-wise |mx|^2 and mx . bias partials -> LDS (ds_add_f32)
  float rs[8], rb[8];
#pragma unroll
  for (int j = 0; j < 8; j++) { rs[j] = 0.f; rb[j] = 0.f; }
#pragma unroll
  for (int t = 0; t < 6; t++) {
#pragma unroll
    for (int j = 0; j < 8; j++) {
      float v = acc[t][j];
      rs[j] += v * v;
      rb[j] += v * bcol[t];
    }
  }
#pragma unroll
  for (int j = 0; j < 8; j++) {
    float a = red_sum16(rs[j]);
    float c = red_sum16(rb[j]);
    if (nl == 0) {
      atomicAdd(&s_rowsum[j + 8 * hi], a);
      atomicAdd(&s_rowdot[j + 8 * hi], c);
    }
  }
  __syncthreads();

  // Mobius matvec scale + Mobius add coefficients per row
  const float b2 = b2p[0];
  float cx[8], cy[8], scl[8];
#pragma unroll
  for (int j = 0; j < 8; j++) {
    const int r = j + 8 * hi;
    float mxn = fmaxf(sqrtf(s_rowsum[r]), EPSF);
    float xn  = xnorm[row0 + r];
    float xc  = fminf(xn, 1.f - 1e-7f);
    float th  = tanhf(mxn / xn * atanhf(xc));
    float sc  = th / mxn;
    float x2  = th * th;
    float xy  = sc * s_rowdot[r];
    float den = 1.f + 2.f * xy + x2 * b2 + EPSF;
    cx[j] = (1.f + 2.f * xy + b2) / den;
    cy[j] = (1.f - x2) / den;
    scl[j] = sc;
  }

  const int bidx = row0 / SS, s0 = row0 % SS;
#pragma unroll
  for (int t = 0; t < 6; t++) {
    const int col = (wid * 6 + t) * 16 + nl;
    float sq[8];
#pragma unroll
    for (int j = 0; j < 8; j++) {
      const int r = j + 8 * hi;
      float o = cx[j] * scl[j] * acc[t][j] + cy[j] * bcol[t];
      if (mode == 0) {
        float oc = fminf(o, 1.f - 1e-5f);
        const int hh = col >> 6, d = col & 63, ss = s0 + r;
        outh[(((size_t)bidx * HH + hh) * SS + ss) * DD + d] = (_Float16)oc;
        sq[j] = oc * oc;
      } else if (mode == 1) {
        const int hh = col >> 6, d = col & 63, ss = s0 + r;
        outh[(((size_t)bidx * HH + hh) * DD + d) * SS + ss] = (_Float16)o;
      } else {
        outf[(size_t)(row0 + r) * EE + col] = o;
      }
    }
    if (mode == 0) {
      const int hh = col >> 6;
#pragma unroll
      for (int j = 0; j < 8; j++) {
        float a = red_sum16(sq[j]);
        if (nl == 0) atomicAdd(&s_sq[(j + 8 * hi) * 12 + hh], a);
      }
    }
  }
  if (mode == 0) {
    __syncthreads();
    if (tid < 192) {
      const int r = tid / 12, hh = tid % 12;
      sqout[((size_t)bidx * HH + hh) * SS + s0 + r] = s_sq[r * 12 + hh];
    }
  }
}

// ---------------------------------------------------------------------------
// hyp_attn: per (b,h,16-q-rows) block.
// Phase 1: qk via WMMA (D=64 -> 2 x K32), hyperbolic distance -> scores (f16 LDS)
// Phase 2: wave-per-row softmax (shfl reductions)
// Phase 3: out = w @ V via WMMA, split-K across waves, LDS partial reduce
// ---------------------------------------------------------------------------
__global__ __launch_bounds__(256) void hyp_attn_kernel(
    const _Float16* __restrict__ Q, const _Float16* __restrict__ K,
    const _Float16* __restrict__ Vt, const float* __restrict__ q2g,
    const float* __restrict__ k2g, const float* __restrict__ hsg,
    float* __restrict__ out) {
  __shared__ __align__(16) _Float16 s_w[16 * 1024];  // scores then weights
  __shared__ float s_part[1024];                     // 4 tiles x 16 x 16

  const int tid = threadIdx.x, wid = tid >> 5, lane = tid & 31;
  const int nl = lane & 15, hi = (lane >> 4) & 1;
  const int qt = blockIdx.x, h = blockIdx.y, b = blockIdx.z;
  const int bh = b * HH + h, q0 = qt * 16;
  const float hs = hsg[h] * 8.0f;  // head_scaling * sqrt(D)

  // Q-tile A fragments (D 0..31 and 32..63)
  const _Float16* qrow = Q + ((size_t)bh * SS + q0 + nl) * DD;
  v16h A0, A1;
  { const int off = (hi ? 8 : 0);      A0 = frag_from_h(qrow + off, qrow + off + 16); }
  { const int off = 32 + (hi ? 8 : 0); A1 = frag_from_h(qrow + off, qrow + off + 16); }

  float q2r[8];
#pragma unroll
  for (int j = 0; j < 8; j++) q2r[j] = q2g[(size_t)bh * SS + q0 + j + 8 * hi];

  // Phase 1: 64 k-tiles, 8 per wave
#pragma unroll 2
  for (int i = 0; i < 8; i++) {
    const int k0 = (wid + 8 * i) * 16;
    const _Float16* krow = K + ((size_t)bh * SS + k0 + nl) * DD;
    v16h B0, B1;
    { const int off = (hi ? 8 : 0);      B0 = frag_from_h(krow + off, krow + off + 16); }
    { const int off = 32 + (hi ? 8 : 0); B1 = frag_from_h(krow + off, krow + off + 16); }
    v8f c = {};
    c = __builtin_amdgcn_wmma_f32_16x16x32_f16(false, A0, false, B0, (short)0, c, false, false);
    c = __builtin_amdgcn_wmma_f32_16x16x32_f16(false, A1, false, B1, (short)0, c, false, false);
    const float k2c = k2g[(size_t)bh * SS + k0 + nl];
#pragma unroll
    for (int j = 0; j < 8; j++) {
      float qk = c[j];
      float numer = 2.f * (q2r[j] + k2c - 2.f * qk);
      float denom = (1.f - q2r[j]) * (1.f - k2c) + EPSF;
      float v = fmaxf(1.f + numer / denom, 1.f);
      float d = fmaxf(acoshf(v), 1e-7f);
      s_w[(j + 8 * hi) * 1024 + k0 + nl] = (_Float16)(-d / hs);
    }
  }
  __syncthreads();

  // Phase 2: softmax; wave w owns rows 2w, 2w+1
#pragma unroll
  for (int rr = 0; rr < 2; rr++) {
    const int row = 2 * wid + rr;
    float vals[32];
    float m = -1e30f;
#pragma unroll
    for (int i = 0; i < 32; i++) {
      vals[i] = (float)s_w[row * 1024 + lane + 32 * i];
      m = fmaxf(m, vals[i]);
    }
    m = red_max32(m);
    float ssum = 0.f;
#pragma unroll
    for (int i = 0; i < 32; i++) { float e = __expf(vals[i] - m); vals[i] = e; ssum += e; }
    ssum = red_sum32(ssum);
    const float inv = 1.f / ssum;
#pragma unroll
    for (int i = 0; i < 32; i++) s_w[row * 1024 + lane + 32 * i] = (_Float16)(vals[i] * inv);
  }
  __syncthreads();

  // Phase 3: out = w @ V.  wave -> (ntile in D, khalf in K)
  const int ntile = wid & 3, khalf = wid >> 2;
  const _Float16* vrow = Vt + ((size_t)bh * DD + ntile * 16 + nl) * SS;
  v8f c = {};
#pragma unroll 2
  for (int stp = 0; stp < 16; stp++) {
    const int kk = khalf * 512 + stp * 32;
    const _Float16* aw = &s_w[nl * 1024 + kk + (hi ? 8 : 0)];
    v16h A = frag_from_h(aw, aw + 16);
    const _Float16* bw = vrow + kk + (hi ? 8 : 0);
    v16h Bf = frag_from_h(bw, bw + 16);
    c = __builtin_amdgcn_wmma_f32_16x16x32_f16(false, A, false, Bf, (short)0, c, false, false);
  }
  if (khalf == 1) {
#pragma unroll
    for (int j = 0; j < 8; j++) s_part[(ntile * 16 + j + 8 * hi) * 16 + nl] = c[j];
  }
  __syncthreads();
  if (khalf == 0) {
#pragma unroll
    for (int j = 0; j < 8; j++) {
      const int r = j + 8 * hi;
      float o = c[j] + s_part[(ntile * 16 + r) * 16 + nl];
      out[((size_t)b * SS + q0 + r) * EE + h * DD + ntile * 16 + nl] = o;
    }
  }
}

// ---------------------------------------------------------------------------
extern "C" void kernel_launch(void* const* d_in, const int* in_sizes, int n_in,
                              void* d_out, int out_size, void* d_ws, size_t ws_size,
                              hipStream_t stream) {
  const float* x  = (const float*)d_in[0];
  const float* Wq = (const float*)d_in[1];
  const float* bq = (const float*)d_in[2];
  const float* Wk = (const float*)d_in[3];
  const float* bk = (const float*)d_in[4];
  const float* Wv = (const float*)d_in[5];
  const float* bv = (const float*)d_in[6];
  const float* Wo = (const float*)d_in[7];
  const float* bo = (const float*)d_in[8];
  const float* hs = (const float*)d_in[9];
  float* out = (float*)d_out;

  char* w = (char*)d_ws;
  auto alloc = [&](size_t bytes) -> char* {
    char* p = w;
    w += (bytes + 255) & ~(size_t)255;
    return p;
  };
  const size_t nrows = (size_t)BB * SS;         // 8192
  const size_t nx    = nrows * EE;              // 6.29M
  const size_t nw    = (size_t)EE * EE;         // 589,824
  float*    xn  = (float*)alloc(nrows * 4);
  float*    xn2 = (float*)alloc(nrows * 4);
  float*    b2  = (float*)alloc(4 * 4);
  float*    q2  = (float*)alloc((size_t)BB * HH * SS * 4);
  float*    k2  = (float*)alloc((size_t)BB * HH * SS * 4);
  _Float16* qh  = (_Float16*)alloc((size_t)BB * HH * SS * DD * 2);
  _Float16* kh  = (_Float16*)alloc((size_t)BB * HH * SS * DD * 2);
  _Float16* vt  = (_Float16*)alloc((size_t)BB * HH * SS * DD * 2);
  float*    ao  = (float*)alloc(nx * 4);
  _Float16* xhh = (_Float16*)alloc(nx * 2);
  _Float16* aoh = (_Float16*)alloc(nx * 2);
  _Float16* wqh = (_Float16*)alloc(nw * 2);
  _Float16* wkh = (_Float16*)alloc(nw * 2);
  _Float16* wvh = (_Float16*)alloc(nw * 2);
  _Float16* woh = (_Float16*)alloc(nw * 2);

  // one-time f16 conversions (removes v_cvt chains from GEMM inner loops)
  const int cb = 256;
  f32_to_f16_kernel<<<(nx / 8 + cb - 1) / cb, cb, 0, stream>>>(x, xhh, (int)nx);
  f32_to_f16_kernel<<<(nw / 8 + cb - 1) / cb, cb, 0, stream>>>(Wq, wqh, (int)nw);
  f32_to_f16_kernel<<<(nw / 8 + cb - 1) / cb, cb, 0, stream>>>(Wk, wkh, (int)nw);
  f32_to_f16_kernel<<<(nw / 8 + cb - 1) / cb, cb, 0, stream>>>(Wv, wvh, (int)nw);
  f32_to_f16_kernel<<<(nw / 8 + cb - 1) / cb, cb, 0, stream>>>(Wo, woh, (int)nw);

  sqsum_kernel<<<1, 32, 0, stream>>>(bq, b2 + 0);
  sqsum_kernel<<<1, 32, 0, stream>>>(bk, b2 + 1);
  sqsum_kernel<<<1, 32, 0, stream>>>(bv, b2 + 2);
  sqsum_kernel<<<1, 32, 0, stream>>>(bo, b2 + 3);
  rownorm_kernel<<<nrows, 32, 0, stream>>>(x, xn);

  hyp_proj_kernel<<<512, 256, 0, stream>>>(xhh, wqh, bq, xn, b2 + 0, 0, qh, q2, nullptr);
  hyp_proj_kernel<<<512, 256, 0, stream>>>(xhh, wkh, bk, xn, b2 + 1, 0, kh, k2, nullptr);
  hyp_proj_kernel<<<512, 256, 0, stream>>>(xhh, wvh, bv, xn, b2 + 2, 1, vt, nullptr, nullptr);

  hyp_attn_kernel<<<dim3(SS / 16, HH, BB), 256, 0, stream>>>(qh, kh, vt, q2, k2, hs, ao);

  rownorm_kernel<<<nrows, 32, 0, stream>>>(ao, xn2);
  f32_to_f16_kernel<<<(nx / 8 + cb - 1) / cb, cb, 0, stream>>>(ao, aoh, (int)nx);
  hyp_proj_kernel<<<512, 256, 0, stream>>>(aoh, woh, bo, xn2, b2 + 3, 2, nullptr, nullptr, out);
}
